// LSTM_63754494542358
// MI455X (gfx1250) — matrix operand order repaired and checked by reading, not compiled, not run on hardware
//
#include <hip/hip_runtime.h>
#include <hip/hip_bf16.h>

// ---------------------------------------------------------------------------
// Bidirectional 2-layer LSTM (SEQ=2048, H=1024) + 32000-way linear head.
//
//  Phase 1: f32->bf16 conversion of x and the 4 w_ih matrices only.
//  Phase 2: xg = x @ w_ih^T via v_wmma_f32_16x16x32_bf16 (f32 accumulate).
//  Phase 3: persistent recurrence: w_hh slice LDS-resident in *fp32*
//           (256KB/WG -- uses CDNA5's 320KB WGP LDS), grid barrier per step,
//           both directions concurrently (blockIdx.y).
//  Phase 4: logits GEMV in fp32 (bandwidth-bound, 262MB @ 23.3TB/s ~ 11us).
// ---------------------------------------------------------------------------

typedef __attribute__((ext_vector_type(16))) __bf16 v16bf;
typedef __attribute__((ext_vector_type(8)))  __bf16 v8bf;
typedef __attribute__((ext_vector_type(8)))  float  v8f;

#define SEQT  2048
#define INSZ  1024
#define HID   1024
#define H4    4096
#define X1SZ  2048
#define OUTN  32000

// ---------------- f32 -> bf16 conversion ----------------
__global__ void k_f32_to_bf16(const float* __restrict__ x, __bf16* __restrict__ y, int n) {
  int i = blockIdx.x * blockDim.x + threadIdx.x;
  int stride = gridDim.x * blockDim.x;
  for (; i < n; i += stride) y[i] = (__bf16)x[i];
}

// ---------------- per-layer state init ----------------
__global__ void k_init_state(const float* __restrict__ h0, const float* __restrict__ c0,
                             int slot0, float* __restrict__ h_state,
                             float* __restrict__ c_state, unsigned* __restrict__ bars) {
  int i = blockIdx.x * blockDim.x + threadIdx.x;
  if (i < 2 * HID) {
    int dir = i >> 10, j = i & (HID - 1);
    h_state[i] = h0[(size_t)(slot0 + dir) * HID + j];
    c_state[i] = c0[(size_t)(slot0 + dir) * HID + j];
  }
  if (i < 8) bars[i] = 0u;
}

// ---------------- WMMA GEMM: C[M,N] = A[M,K] * B[N,K]^T (bf16 in, f32 out) --
// One 16x16 tile per wave; 8 waves/WG arranged as a 32x64 supertile.
__global__ void k_gemm_bf16_wmma(const __bf16* __restrict__ A, const __bf16* __restrict__ B,
                                 float* __restrict__ C, int M, int N, int K) {
  const int lane = threadIdx.x & 31;
  const int wave = threadIdx.x >> 5;
  const int sn_tiles = N >> 6;
  const int sm = blockIdx.x / sn_tiles;
  const int sn = blockIdx.x % sn_tiles;
  const int m0 = sm * 32 + (wave >> 2) * 16;
  const int n0 = sn * 64 + (wave & 3) * 16;
  if (m0 >= M) return;

  const int r  = lane & 15;   // row within tile (A: M-row, B: N-row of W)
  const int hi = lane >> 4;   // lane half selects K sub-range

  // A frag (16x32): lane holds A[m0+r][k0 + hi*8 + 0..7] and [... +16 ..+7]
  const __bf16* pa = A + (size_t)(m0 + r) * K + hi * 8;
  // B frag (32x16 = W^T): lane holds W[n0+r][k0 + hi*8 + 0..7] and [... +16 ..+7]
  const __bf16* pb = B + (size_t)(n0 + r) * K + hi * 8;

  v8f acc = {};
  for (int k0 = 0; k0 < K; k0 += 32) {
    union { v16bf v; v8bf h[2]; } ua, ub;
    ua.h[0] = *(const v8bf*)(pa + k0);
    ua.h[1] = *(const v8bf*)(pa + k0 + 16);
    ub.h[0] = *(const v8bf*)(pb + k0);
    ub.h[1] = *(const v8bf*)(pb + k0 + 16);
    acc = __builtin_amdgcn_wmma_f32_16x16x32_bf16(false, ua.v, false, ub.v,
                                                  (short)0, acc, false, false);
  }
  // C/D layout: VGPR v -> row m0 + hi*8 + v, col n0 + r
  float* c = C + (size_t)(m0 + hi * 8) * N + n0 + r;
#pragma unroll
  for (int v = 0; v < 8; ++v) c[(size_t)v * N] = acc[v];
}

// ---------------- device-wide barrier (per direction) ----------------
__device__ __forceinline__ void grid_barrier(unsigned* cnt, unsigned* gen, unsigned total) {
  __threadfence();
  __syncthreads();
  if (threadIdx.x == 0) {
    unsigned g = __atomic_load_n(gen, __ATOMIC_RELAXED);
    unsigned prev = atomicAdd(cnt, 1u);
    if (prev == total - 1u) {
      __atomic_store_n(cnt, 0u, __ATOMIC_RELAXED);
      __atomic_fetch_add(gen, 1u, __ATOMIC_RELEASE);
    } else {
      while (__atomic_load_n(gen, __ATOMIC_ACQUIRE) == g) __builtin_amdgcn_s_sleep(2);
    }
  }
  __syncthreads();
  __threadfence();
}

__device__ __forceinline__ float sigmoidf_(float x) { return 1.0f / (1.0f + expf(-x)); }

// ---------------- persistent LSTM recurrence ----------------
// grid (HID/16, 2): blockIdx.x = 16-wide h slice, blockIdx.y = direction.
// 512 threads. LDS: 64 rows x 1024 f32 of w_hh (256KB) + h(4KB) + g(256B) + c(64B).
__global__ void k_lstm_rec(const float* __restrict__ xg_f, const float* __restrict__ xg_b,
                           const float* __restrict__ whh_f, const float* __restrict__ whh_b,
                           float* __restrict__ h_state, float* __restrict__ c_state,
                           __bf16* seq_out, float* last_h,
                           float* __restrict__ hn_out, float* __restrict__ cn_out,
                           int slot0, unsigned* __restrict__ bars, int T) {
  extern __shared__ char smem[];
  float* lds_w = (float*)smem;                                  // 64*1024 f32
  float* lds_h = (float*)(smem + 64 * 1024 * sizeof(float));    // 1024 f32
  float* lds_g = lds_h + 1024;                                  // 64 f32
  float* lds_c = lds_g + 64;                                    // 16 f32

  const int dir = blockIdx.y;
  const int j0  = blockIdx.x * 16;
  const int tid = threadIdx.x;
  const float* xg  = dir ? xg_b : xg_f;
  const float* whh = dir ? whh_b : whh_f;
  unsigned* cnt = bars + dir * 2;
  unsigned* gen = cnt + 1;

  // Stage the owned w_hh rows (gate*H + j0 + jj, jj=0..15) into LDS once (f32).
  for (int q = tid; q < 16384; q += 512) {          // 16384 x 16B chunks
    int rr = q >> 8;                                // 0..63  (gate*16 + jj)
    int kk = (q & 255) << 2;
    int grow = ((rr >> 4) << 10) + j0 + (rr & 15);  // gate*1024 + j
    *(float4*)(lds_w + (size_t)rr * 1024 + kk) =
        *(const float4*)(whh + (size_t)grow * HID + kk);
  }
  if (tid < 16) lds_c[tid] = c_state[dir * HID + j0 + tid];
  __syncthreads();

  const int rrow = tid >> 3;   // 0..63 : gate-row within slice
  const int oct  = tid & 7;    // 128-element segment of the dot product

  for (int s = 0; s < T; ++s) {
    const int t = dir ? (T - 1 - s) : s;

    // broadcast h through LDS
    for (int i = tid; i < HID; i += 512) lds_h[i] = h_state[dir * HID + i];
    __syncthreads();

    // partial dot products: f32 weights (LDS) x f32 h (LDS), b128 LDS loads
    const float4* wr = (const float4*)(lds_w + (size_t)rrow * 1024 + (oct << 7));
    const float4* hr = (const float4*)(lds_h + (oct << 7));
    float sum = 0.f;
#pragma unroll
    for (int k = 0; k < 32; ++k) {
      float4 w = wr[k];
      float4 h = hr[k];
      sum += w.x * h.x + w.y * h.y + w.z * h.z + w.w * h.w;
    }
    sum += __shfl_xor(sum, 1);
    sum += __shfl_xor(sum, 2);
    sum += __shfl_xor(sum, 4);
    if (oct == 0) lds_g[rrow] = sum;
    __syncthreads();

    // gate math + state update for the 16 owned h indices
    if (tid < 16) {
      const int j = j0 + tid;
      const float* xr = xg + (size_t)t * H4;
      float gi = xr[j]           + lds_g[tid];
      float gf = xr[HID + j]     + lds_g[16 + tid];
      float gg = xr[2 * HID + j] + lds_g[32 + tid];
      float go = xr[3 * HID + j] + lds_g[48 + tid];
      float iv = sigmoidf_(gi);
      float fv = sigmoidf_(gf);
      float gv = tanhf(gg);
      float ov = sigmoidf_(go);
      float cv = fv * lds_c[tid] + iv * gv;
      lds_c[tid] = cv;
      float hv = ov * tanhf(cv);
      h_state[dir * HID + j] = hv;
      if (seq_out) seq_out[(size_t)t * X1SZ + dir * HID + j] = (__bf16)hv;
      if (last_h && t == T - 1) last_h[dir * HID + j] = hv;
      if (s == T - 1) {
        hn_out[(size_t)(slot0 + dir) * HID + j] = hv;
        cn_out[(size_t)(slot0 + dir) * HID + j] = cv;
      }
    }
    grid_barrier(cnt, gen, gridDim.x);
  }
}

// ---------------- fp32 GEMV head: logits = h_last @ w_lin^T ----------------
__global__ void k_gemv_logits(const float* __restrict__ W, const float* __restrict__ h,
                              float* __restrict__ out, int N, int K) {
  __shared__ float sh[X1SZ];
  for (int i = threadIdx.x; i < K; i += blockDim.x) sh[i] = h[i];
  __syncthreads();
  const int lane = threadIdx.x & 31;
  const int row  = blockIdx.x * (blockDim.x >> 5) + (threadIdx.x >> 5);
  if (row >= N) return;
  const float* wr = W + (size_t)row * K;
  float sum = 0.f;
  for (int k = lane; k < K; k += 32) sum += wr[k] * sh[k];
  sum += __shfl_xor(sum, 16);
  sum += __shfl_xor(sum, 8);
  sum += __shfl_xor(sum, 4);
  sum += __shfl_xor(sum, 2);
  sum += __shfl_xor(sum, 1);
  if (lane == 0) out[row] = sum;
}

// ---------------------------------------------------------------------------
extern "C" void kernel_launch(void* const* d_in, const int* in_sizes, int n_in,
                              void* d_out, int out_size, void* d_ws, size_t ws_size,
                              hipStream_t stream) {
  (void)in_sizes; (void)n_in; (void)out_size; (void)ws_size;
  const float* x     = (const float*)d_in[0];
  const float* h0    = (const float*)d_in[1];
  const float* c0    = (const float*)d_in[2];
  const float* wih0f = (const float*)d_in[3];
  const float* whh0f = (const float*)d_in[4];
  const float* wih0b = (const float*)d_in[5];
  const float* whh0b = (const float*)d_in[6];
  const float* wih1f = (const float*)d_in[7];
  const float* whh1f = (const float*)d_in[8];
  const float* wih1b = (const float*)d_in[9];
  const float* whh1b = (const float*)d_in[10];
  const float* wlin  = (const float*)d_in[11];

  char* ws = (char*)d_ws;
  size_t off = 0;
  auto take = [&](size_t bytes) -> char* {
    char* p = ws + off;
    off += (bytes + 255) & ~(size_t)255;
    return p;
  };
  __bf16* xbf    = (__bf16*)take((size_t)SEQT * INSZ * 2);
  __bf16* bwih0f = (__bf16*)take((size_t)H4 * INSZ * 2);
  __bf16* bwih0b = (__bf16*)take((size_t)H4 * INSZ * 2);
  __bf16* bwih1f = (__bf16*)take((size_t)H4 * X1SZ * 2);
  __bf16* bwih1b = (__bf16*)take((size_t)H4 * X1SZ * 2);
  float*  xgf    = (float*)take((size_t)SEQT * H4 * 4);
  float*  xgb    = (float*)take((size_t)SEQT * H4 * 4);
  __bf16* x1bf   = (__bf16*)take((size_t)SEQT * X1SZ * 2);
  float*  hstate = (float*)take((size_t)2 * HID * 4);
  float*  cstate = (float*)take((size_t)2 * HID * 4);
  float*  lasth  = (float*)take((size_t)X1SZ * 4);
  unsigned* bars = (unsigned*)take(256);

  float* logits = (float*)d_out;
  float* hn     = logits + OUTN;
  float* cn     = hn + 4 * HID;

  auto cvt = [&](const float* src, __bf16* dst, int n) {
    k_f32_to_bf16<<<2048, 256, 0, stream>>>(src, dst, n);
  };
  cvt(x,     xbf,    SEQT * INSZ);
  cvt(wih0f, bwih0f, H4 * INSZ);
  cvt(wih0b, bwih0b, H4 * INSZ);
  cvt(wih1f, bwih1f, H4 * X1SZ);
  cvt(wih1b, bwih1b, H4 * X1SZ);

  const int gemm_grid = (SEQT / 32) * (H4 / 64);
  const size_t rec_lds = (size_t)64 * 1024 * 4 + (1024 + 64 + 16) * 4;

  // ---- layer 0 ----
  k_init_state<<<8, 256, 0, stream>>>(h0, c0, 0, hstate, cstate, bars);
  k_gemm_bf16_wmma<<<gemm_grid, 256, 0, stream>>>(xbf, bwih0f, xgf, SEQT, H4, INSZ);
  k_gemm_bf16_wmma<<<gemm_grid, 256, 0, stream>>>(xbf, bwih0b, xgb, SEQT, H4, INSZ);
  k_lstm_rec<<<dim3(HID / 16, 2), 512, rec_lds, stream>>>(
      xgf, xgb, whh0f, whh0b, hstate, cstate, x1bf, nullptr, hn, cn, 0, bars, SEQT);

  // ---- layer 1 ----
  k_init_state<<<8, 256, 0, stream>>>(h0, c0, 2, hstate, cstate, bars);
  k_gemm_bf16_wmma<<<gemm_grid, 256, 0, stream>>>(x1bf, bwih1f, xgf, SEQT, H4, X1SZ);
  k_gemm_bf16_wmma<<<gemm_grid, 256, 0, stream>>>(x1bf, bwih1b, xgb, SEQT, H4, X1SZ);
  k_lstm_rec<<<dim3(HID / 16, 2), 512, rec_lds, stream>>>(
      xgf, xgb, whh1f, whh1b, hstate, cstate, nullptr, lasth, hn, cn, 2, bars, SEQT);

  // ---- head ----
  k_gemv_logits<<<(OUTN + 7) / 8, 256, 0, stream>>>(wlin, lasth, logits, OUTN, X1SZ);
}
